// PostureRNN_71184787964479
// MI455X (gfx1250) — matrix operand order repaired and checked by reading, not compiled, not run on hardware
//
#include <hip/hip_runtime.h>
#include <hip/hip_bf16.h>

// ---- geometry ----
#define HSZ   64       // hidden size
#define TLEN  2048     // sequence length
#define NBAT  1024     // batch
#define BT    16       // batch rows per workgroup (WMMA M)
#define TC    128      // timesteps staged per chunk
#define OSZ   3        // output size

// ---- padded LDS row strides ----
#define XP    20       // x stage row: 20 floats = 80B (16B aligned)
#define HP    72       // h buffer row: 72 bf16 = 144B (16B aligned, bank-clean A loads)

typedef __attribute__((ext_vector_type(16))) __bf16 v16bf;
typedef __attribute__((ext_vector_type(8)))  __bf16 v8bf;
typedef __attribute__((ext_vector_type(8)))  float  v8f;
typedef __attribute__((ext_vector_type(4)))  float  f32x4;

static __device__ __forceinline__ __bf16 f2bf(float f) { return (__bf16)f; }

static __device__ __forceinline__ float fast_tanh(float x) {
#if __has_builtin(__builtin_amdgcn_tanhf)
    return __builtin_amdgcn_tanhf(x);          // native v_tanh_f32 (confirmed)
#else
    return tanhf(x);
#endif
}

// CDNA5 split workgroup barrier (signal early, wait late) with proper
// LDS release/acquire fencing; falls back to __syncthreads.
#if __has_builtin(__builtin_amdgcn_s_barrier_signal) && \
    __has_builtin(__builtin_amdgcn_s_barrier_wait)
static __device__ __forceinline__ void bar_signal() {
    __builtin_amdgcn_fence(__ATOMIC_RELEASE, "workgroup");
    __builtin_amdgcn_s_barrier_signal(-1);
}
static __device__ __forceinline__ void bar_wait() {
    __builtin_amdgcn_s_barrier_wait((short)-1);
    __builtin_amdgcn_fence(__ATOMIC_ACQUIRE, "workgroup");
}
#else
static __device__ __forceinline__ void bar_signal() { __syncthreads(); }
static __device__ __forceinline__ void bar_wait()   {}
#endif

__launch_bounds__(128, 1)
__global__ void lstm_scan_kernel(const float* __restrict__ x,
                                 const float* __restrict__ W_ih,
                                 const float* __restrict__ W_hh,
                                 const float* __restrict__ b_ih,
                                 const float* __restrict__ b_hh,
                                 const float* __restrict__ W_fc,
                                 const float* __restrict__ b_fc,
                                 float* __restrict__ out)
{
    __shared__ float  xs[TC][XP];           // staged x chunk: [t_local][batch_row]
    __shared__ __bf16 hbuf[2][BT][HP];      // ping-pong h state, bf16 [m][k]

    const int tid  = threadIdx.x;
    const int lane = tid & 31;
    const int wave = tid >> 5;           // wave owns hidden columns [16*wave, +16)
    const int hl   = lane >> 4;          // lane half (A/B layout K-half select)
    const int lm   = lane & 15;          // lane mod 16 (M for A, N for B/C)
    const int bt0  = blockIdx.x * BT;    // first batch row of this tile
    const int jcol = wave * 16 + lm;     // this lane's hidden column

    // ------------------------------------------------------------------
    // Resident B fragments. Tile t == gate t (i,f,g,o); tile covers gate
    // columns g = 64*t + 16*wave + n  (n = lane%16).
    // Sigmoid gates (t != 2) pre-scaled by 0.5 (exact in bf16) so
    // sigmoid(z) = 0.5*tanh(z/2) + 0.5  ->  1 v_tanh + 1 pk_fma.
    // B (K=32 x N=16 bf16): element e -> K = kbase + 8*hl + e (e<8),
    // K = kbase + 16 + 8*hl + (e-8) (e>=8).  B[k][n] = W_hh[g][k].
    // ------------------------------------------------------------------
    v16bf bfrag[4][2];
    float wih[4], bias[4];
    #pragma unroll
    for (int t = 0; t < 4; ++t) {
        const float sc = (t == 2) ? 1.f : 0.5f;
        const int g = t * 64 + wave * 16 + lm;       // W_hh row (gate index)
        #pragma unroll
        for (int c = 0; c < 2; ++c) {
            const int kb = c * 32 + hl * 8;
            v16bf bv;
            #pragma unroll
            for (int e = 0; e < 16; ++e)
                bv[e] = f2bf(W_hh[g * HSZ + kb + e + (e >= 8 ? 8 : 0)] * sc);
            bfrag[t][c] = bv;
        }
        wih[t]  = W_ih[g] * sc;
        bias[t] = (b_ih[g] + b_hh[g]) * sc;
    }

    // zero h state read at step 0 (buffer 0); buffer 1 is written first.
    for (int i = tid; i < BT * HP; i += 128) ((__bf16*)hbuf[0])[i] = f2bf(0.f);

    // c state lives in registers: lane owns cells (m = v + 8*hl, j = jcol)
    float creg[8];
    #pragma unroll
    for (int v = 0; v < 8; ++v) creg[v] = 0.f;

    __syncthreads();

    for (int chunk = 0; chunk < TLEN / TC; ++chunk) {
        // ---- stage x chunk (coalesced over t), prefetch next chunk ----
        #pragma unroll 4
        for (int r = 0; r < BT; ++r)
            xs[tid][r] = x[(size_t)(bt0 + r) * TLEN + chunk * TC + tid];
        if (chunk + 1 < TLEN / TC) {
            #pragma unroll 4
            for (int r = 0; r < BT; ++r)
                __builtin_prefetch(&x[(size_t)(bt0 + r) * TLEN + (chunk + 1) * TC + tid], 0, 1);
        }
        __syncthreads();   // closes chunk: xs ready, prior h stores visible

        for (int tl = 0; tl < TC; ++tl) {
            const int rb = tl & 1;            // read buffer (0 at step 0 ✓)

            // ---- h-independent work first: x values + gate C-init ----
            const f32x4 xa = *(const f32x4*)&xs[tl][hl * 8];
            const f32x4 xb = *(const f32x4*)&xs[tl][hl * 8 + 4];
            v8f accs[4];
            #pragma unroll
            for (int t = 0; t < 4; ++t) {
                #pragma unroll
                for (int v = 0; v < 8; ++v) {
                    const float xv = (v < 4) ? xa[v] : xb[v - 4];
                    accs[t][v] = xv * wih[t] + bias[t];   // x*W_ih + bias
                }
            }

            // ---- wait for producers of hbuf[rb] (skip: step 0 covered
            //      by the chunk __syncthreads) ----
            if (tl > 0) bar_wait();

            // ---- A fragments (16x32 bf16): lane m=lm; K runs of 8 at
            //      {8*hl, 16+8*hl} (+32 for chunk 1); one b128 load each.
            const __bf16* hrow = &hbuf[rb][lm][0];
            const int klo = hl * 8;
            v16bf a0, a1;
            *((v8bf*)&a0)       = *(const v8bf*)(hrow + klo);
            *(((v8bf*)&a0) + 1) = *(const v8bf*)(hrow + 16 + klo);
            *((v8bf*)&a1)       = *(const v8bf*)(hrow + 32 + klo);
            *(((v8bf*)&a1) + 1) = *(const v8bf*)(hrow + 48 + klo);

            #pragma unroll
            for (int t = 0; t < 4; ++t) {
                accs[t] = __builtin_amdgcn_wmma_f32_16x16x32_bf16(
                    false, a0, false, bfrag[t][0], (short)0, accs[t], false, false);
                accs[t] = __builtin_amdgcn_wmma_f32_16x16x32_bf16(
                    false, a1, false, bfrag[t][1], (short)0, accs[t], false, false);
            }

            // ---- in-register activation + c/h update ----
            #pragma unroll
            for (int v = 0; v < 8; ++v) {
                const float ig = fast_tanh(accs[0][v]) * 0.5f + 0.5f; // sigmoid
                const float fg = fast_tanh(accs[1][v]) * 0.5f + 0.5f; // sigmoid
                const float gg = fast_tanh(accs[2][v]);               // tanh
                const float og = fast_tanh(accs[3][v]) * 0.5f + 0.5f; // sigmoid
                const float c  = fg * creg[v] + ig * gg;
                creg[v] = c;
                hbuf[rb ^ 1][v + 8 * hl][jcol] = f2bf(og * fast_tanh(c));
            }

            // signal h[rb^1] complete (last step: chunk __syncthreads /
            // final __syncthreads below closes instead)
            if (tl < TC - 1) bar_signal();
        }
    }
    __syncthreads();   // final h stores visible for epilogue

    // ---- epilogue: out = h_last @ W_fc^T + b_fc (16x64 x 64x3) ----
    // last global step has tl = TC-1 (odd) -> wrote buffer 0
    if (tid < BT) {
        const int m = tid;
        #pragma unroll
        for (int o = 0; o < OSZ; ++o) {
            float s = b_fc[o];
            #pragma unroll 8
            for (int k = 0; k < HSZ; ++k)
                s += (float)hbuf[0][m][k] * W_fc[o * HSZ + k];
            out[(size_t)(bt0 + m) * OSZ + o] = s;
        }
    }
}

extern "C" void kernel_launch(void* const* d_in, const int* in_sizes, int n_in,
                              void* d_out, int out_size, void* d_ws, size_t ws_size,
                              hipStream_t stream) {
    const float* x    = (const float*)d_in[0];
    const float* W_ih = (const float*)d_in[1];
    const float* W_hh = (const float*)d_in[2];
    const float* b_ih = (const float*)d_in[3];
    const float* b_hh = (const float*)d_in[4];
    const float* W_fc = (const float*)d_in[5];
    const float* b_fc = (const float*)d_in[6];
    float* out = (float*)d_out;

    dim3 grid(NBAT / BT);   // 64 workgroups, one batch tile each
    dim3 block(128);        // 4 waves; each owns a 16-wide hidden slice
    lstm_scan_kernel<<<grid, block, 0, stream>>>(x, W_ih, W_hh, b_ih, b_hh,
                                                 W_fc, b_fc, out);
}
